// SO3RotInvPointTransformer_79783312490949
// MI455X (gfx1250) — matrix operand order repaired and checked by reading, not compiled, not run on hardware
//
#include <hip/hip_runtime.h>
#include <hip/hip_bf16.h>

typedef __attribute__((ext_vector_type(16))) _Float16 v16h;
typedef __attribute__((ext_vector_type(8)))  float    v8f;
typedef __attribute__((ext_vector_type(4)))  unsigned int v4u;
typedef __attribute__((ext_vector_type(8)))  int      v8i;
typedef __attribute__((ext_vector_type(4)))  int      v4i;

#define BB 4
#define NN 2048
#define DD 128
#define HH 4
#define ROWS (BB*NN)         // 8192
#define RAD2 0.36f
#define NEPS2 1e-24f
#define LNEPS 1e-5f

#if defined(__gfx1250__) && __has_builtin(__builtin_amdgcn_tensor_load_to_lds)
#define USE_TDM 1
#else
#define USE_TDM 0
#endif

union F8  { float4 f[2]; float s[8];  };
union F16F{ float4 f[4]; float s[16]; };
union H8  { uint4  u;    _Float16 h[8]; };

__device__ __forceinline__ v16h load_v16h_g(const _Float16* p) {
    union { uint4 u[2]; v16h v; } t;
    const uint4* q = (const uint4*)p;
    t.u[0] = q[0]; t.u[1] = q[1];
    return t.v;
}

// ---------------------------------------------------------------------------
// Generic 128-K GEMM tile kernel: C(16x16) = A(rows x 128) * W^T(:,16) + bias
// One wave per 16x16 tile, K-loop of 4 x v_wmma_f32_16x16x32_f16.
// TRANS_OUT: store f16 transposed [col][row]   (for V^T, B-operand friendly)
// else row-major, optional residual add, f16 or f32 output.
// ---------------------------------------------------------------------------
template<bool A_F16, bool TRANS_OUT, bool ADD_RES, bool OUT_F16>
__global__ __launch_bounds__(128) void gemm_wmma(
    const void* __restrict__ Ain, const float* __restrict__ W,
    const float* __restrict__ bias, const float* __restrict__ res,
    void* __restrict__ Out)
{
    const int lane = threadIdx.x & 31;
    const int wv   = threadIdx.x >> 5;
    const int hi   = (lane >> 4) & 1;
    const int nl   = lane & 15;
    const int tile = blockIdx.x * 4 + wv;        // 4096 tiles total
    const int r0   = (tile >> 3) * 16;           // 512 row tiles
    const int c0   = (tile & 7) * 16;            // 8 col tiles

    v8f c;
    const float bvv = bias[c0 + nl];
    #pragma unroll
    for (int r = 0; r < 8; ++r) c[r] = bvv;

    const int arow = r0 + nl;                    // A row = M = lane%16
    #pragma unroll
    for (int kk = 0; kk < DD; kk += 32) {
        v16h a, b;
        if (A_F16) {
            const _Float16* A  = (const _Float16*)Ain;
            const _Float16* p0 = A + (size_t)arow * DD + kk + hi * 8;
            H8 t0, t1;
            t0.u = *(const uint4*)p0;
            t1.u = *(const uint4*)(p0 + 16);
            #pragma unroll
            for (int e = 0; e < 8; ++e) { a[e] = t0.h[e]; a[e + 8] = t1.h[e]; }
        } else {
            const float* A  = (const float*)Ain;
            const float* p0 = A + (size_t)arow * DD + kk + hi * 8;
            F8 g0, g1;
            g0.f[0] = *(const float4*)(p0);      g0.f[1] = *(const float4*)(p0 + 4);
            g1.f[0] = *(const float4*)(p0 + 16); g1.f[1] = *(const float4*)(p0 + 20);
            #pragma unroll
            for (int e = 0; e < 8; ++e) { a[e] = (_Float16)g0.s[e]; a[e + 8] = (_Float16)g1.s[e]; }
        }
        // B[k][o] = W[o][k]; lane holds col o = c0+nl, K = kk + hi*8 + {0..7, 16..23}
        const float* wp = W + (size_t)(c0 + nl) * DD + kk + hi * 8;
        F16F wb;
        wb.f[0] = *(const float4*)(wp);     wb.f[1] = *(const float4*)(wp + 4);
        wb.f[2] = *(const float4*)(wp + 16); wb.f[3] = *(const float4*)(wp + 20);
        #pragma unroll
        for (int e = 0; e < 16; ++e) b[e] = (_Float16)wb.s[e];

        c = __builtin_amdgcn_wmma_f32_16x16x32_f16(false, a, false, b,
                                                   (short)0, c, false, false);
    }

    if (TRANS_OUT) {
        _Float16* O = (_Float16*)Out;            // [DD][ROWS] f16
        H8 t;
        #pragma unroll
        for (int r = 0; r < 8; ++r) t.h[r] = (_Float16)c[r];
        *(uint4*)(O + (size_t)(c0 + nl) * ROWS + r0 + hi * 8) = t.u;
    } else {
        #pragma unroll
        for (int r = 0; r < 8; ++r) {
            const int row = r0 + hi * 8 + r;
            const size_t idx = (size_t)row * DD + c0 + nl;
            float v = c[r];
            if (ADD_RES) v += res[idx];
            if (OUT_F16) ((_Float16*)Out)[idx] = (_Float16)v;
            else         ((float*)Out)[idx]    = v;
        }
    }
}

// ---------------------------------------------------------------------------
// Fused radius-attention kernel. Block = 16 queries (one batch), 4 waves
// split the 64 key tiles of 32. Lane L owns query L%16 and exactly the
// WMMA A-operand K-slots (K = 8*ch + e + 16*hi), so softmax weights pack
// straight into A registers. Per key tile the wave issues one TDM
// tensor_load_to_lds for its 128x32 f16 V^T slab, overlaps the DMA with the
// 3->32->4 pairwise MLP, then s_wait_tensorcnt gates 8 WMMAs whose B
// operands are contiguous LDS reads.
// ---------------------------------------------------------------------------
__global__ __launch_bounds__(128) void attn_kernel(
    const float* __restrict__ coords, const float* __restrict__ dirs,
    const float* __restrict__ w1, const float* __restrict__ b1,
    const float* __restrict__ w2, const float* __restrict__ b2,
    const _Float16* __restrict__ Vt, _Float16* __restrict__ attnout)
{
    __shared__ float wpk[32][8];                 // {w1x,w1y,w1z,b1, w2_h0..h3}
    __shared__ float qd[16][8];                  // {cx,cy,cz,cn, dx,dy,dz,-}
    __shared__ float md[4][32][8];               // per-wave key-tile staging
    // 32 KB region with two disjoint lifetimes:
    //   during key loop : per-wave V^T tiles [4][128][32] f16 (TDM dest)
    //   after key loop  : cross-wave C partials [4][HH][2][16][16] f32
    __shared__ __align__(16) char ovl[4 * HH * 2 * 16 * 16 * 4];
    __shared__ float sumW[4][2][HH][16];         // per-wave,half exp-sums

    _Float16 (*vtile)[DD][32]      = (_Float16 (*)[DD][32])ovl;
    float (*accW)[HH][2][16][16]   = (float (*)[HH][2][16][16])ovl;

    const int tid  = threadIdx.x;
    const int lane = tid & 31, wv = tid >> 5;
    const int hi   = (lane >> 4) & 1, nl = lane & 15;
    const int b    = blockIdx.x >> 7;            // 128 query tiles per batch
    const int q0   = (blockIdx.x & 127) * 16;

    if (tid < 32) {
        wpk[tid][0] = w1[tid * 3 + 0];
        wpk[tid][1] = w1[tid * 3 + 1];
        wpk[tid][2] = w1[tid * 3 + 2];
        wpk[tid][3] = b1[tid];
        wpk[tid][4] = w2[0 * 32 + tid];
        wpk[tid][5] = w2[1 * 32 + tid];
        wpk[tid][6] = w2[2 * 32 + tid];
        wpk[tid][7] = w2[3 * 32 + tid];
    }
    if (tid < 16) {
        const int n = q0 + tid;
        const float cx = coords[(size_t)(b * NN + n) * 3 + 0];
        const float cy = coords[(size_t)(b * NN + n) * 3 + 1];
        const float cz = coords[(size_t)(b * NN + n) * 3 + 2];
        qd[tid][0] = cx; qd[tid][1] = cy; qd[tid][2] = cz;
        qd[tid][3] = cx * cx + cy * cy + cz * cz;
        qd[tid][4] = dirs[(size_t)(b * NN + n) * 3 + 0];
        qd[tid][5] = dirs[(size_t)(b * NN + n) * 3 + 1];
        qd[tid][6] = dirs[(size_t)(b * NN + n) * 3 + 2];
        qd[tid][7] = 0.f;
    }
    __syncthreads();

    const float qx = qd[nl][0], qy = qd[nl][1], qz = qd[nl][2], qcn = qd[nl][3];
    const float qdx = qd[nl][4], qdy = qd[nl][5], qdz = qd[nl][6];
    const float b2r[4] = { b2[0], b2[1], b2[2], b2[3] };

    v8f acc[HH][2];
    #pragma unroll
    for (int h = 0; h < HH; ++h)
        #pragma unroll
        for (int dt = 0; dt < 2; ++dt)
            #pragma unroll
            for (int r = 0; r < 8; ++r) acc[h][dt][r] = 0.f;
    float sm[HH] = {0.f, 0.f, 0.f, 0.f};

    for (int it = 0; it < NN / 128; ++it) {      // 16 key tiles per wave
        const int mbase = (it * 4 + wv) * 32;
        __syncthreads();                         // md + vtile reuse fence

#if USE_TDM
        {   // One TDM op: V^T slab [128 dims][32 keys] f16 -> vtile[wv].
            // D# group0: count=1 | lds_addr | global_addr(57b) | type=2.
            const unsigned lds_off = (unsigned)(uintptr_t)&vtile[wv][0][0];
            const unsigned long long ga =
                (unsigned long long)(uintptr_t)(Vt + (size_t)b * NN + mbase);
            v4u g0;
            g0[0] = 1u;
            g0[1] = lds_off;
            g0[2] = (unsigned)(ga & 0xffffffffu);
            g0[3] = (unsigned)((ga >> 32) & 0x01ffffffu) | (2u << 30);
            // D# group1: data_size=2B; tensor 8192x128, tile 32x128,
            // dim0 stride 8192 elements; no mask/pad/iterate.
            v8i g1;
            g1[0] = 0x00010000;                        // data_size=1 (2 bytes)
            g1[1] = (int)((ROWS & 0xffff) << 16);      // tensor_dim0[15:0]
            g1[2] = (int)((ROWS >> 16) | (DD << 16));  // dim0 hi | tensor_dim1 lo
            g1[3] = (int)(32 << 16);                   // dim1 hi=0 | tile_dim0=32
            g1[4] = DD;                                // tile_dim1=128, tile_dim2=0
            g1[5] = ROWS;                              // tensor_dim0_stride[31:0]
            g1[6] = 0;                                 // stride hi | dim1_stride lo
            g1[7] = 0;
            v4i gz = {0, 0, 0, 0};
#if __clang_major__ >= 23
            v8i gz8 = {0, 0, 0, 0, 0, 0, 0, 0};
            __builtin_amdgcn_tensor_load_to_lds(g0, g1, gz, gz, gz8, 0);
#else
            __builtin_amdgcn_tensor_load_to_lds(g0, g1, gz, gz, 0);
#endif
        }
#endif
        {
            const int m = mbase + lane;
            const float cx = coords[(size_t)(b * NN + m) * 3 + 0];
            const float cy = coords[(size_t)(b * NN + m) * 3 + 1];
            const float cz = coords[(size_t)(b * NN + m) * 3 + 2];
            md[wv][lane][0] = cx; md[wv][lane][1] = cy; md[wv][lane][2] = cz;
            md[wv][lane][3] = cx * cx + cy * cy + cz * cz;
            md[wv][lane][4] = dirs[(size_t)(b * NN + m) * 3 + 0];
            md[wv][lane][5] = dirs[(size_t)(b * NN + m) * 3 + 1];
            md[wv][lane][6] = dirs[(size_t)(b * NN + m) * 3 + 2];
            md[wv][lane][7] = 0.f;
            if (it + 1 < NN / 128)               // pull next key tile toward L2/L0
                __builtin_prefetch(&coords[(size_t)(b * NN + mbase + 128 + lane) * 3], 0, 1);
        }
        __syncthreads();

        v16h a[HH];
        #pragma unroll
        for (int ch = 0; ch < 2; ++ch) {         // 8 keys per chunk per lane
            float rx[8], ry[8], rz[8];
            float l0[8], l1[8], l2[8], l3[8];
            unsigned mbits = 0;
            #pragma unroll
            for (int e = 0; e < 8; ++e) {
                const int K = ch * 8 + e + hi * 16;   // WMMA A K-slot
                const float4 m0 = *(const float4*)&md[wv][K][0];
                const float4 m1 = *(const float4*)&md[wv][K][4];
                const float d2 = qcn + m0.w - 2.f * (qx * m0.x + qy * m0.y + qz * m0.z);
                if (d2 < RAD2) mbits |= (1u << e);
                const float ax = qdx - m1.x, ay = qdy - m1.y, az = qdz - m1.z;
                const float inv = rsqrtf(fmaxf(ax * ax + ay * ay + az * az, NEPS2));
                rx[e] = ax * inv; ry[e] = ay * inv; rz[e] = az * inv;
                l0[e] = b2r[0]; l1[e] = b2r[1]; l2[e] = b2r[2]; l3[e] = b2r[3];
            }
            #pragma unroll 4
            for (int i = 0; i < 32; ++i) {       // 3->32->4 MLP, weights in LDS
                const float4 wa = *(const float4*)&wpk[i][0];
                const float4 wb = *(const float4*)&wpk[i][4];
                #pragma unroll
                for (int e = 0; e < 8; ++e) {
                    float hv = fmaf(wa.x, rx[e], fmaf(wa.y, ry[e], fmaf(wa.z, rz[e], wa.w)));
                    hv = fmaxf(hv, 0.f);
                    l0[e] = fmaf(hv, wb.x, l0[e]);
                    l1[e] = fmaf(hv, wb.y, l1[e]);
                    l2[e] = fmaf(hv, wb.z, l2[e]);
                    l3[e] = fmaf(hv, wb.w, l3[e]);
                }
            }
            #pragma unroll
            for (int e = 0; e < 8; ++e) {        // masked exp -> A registers
                const bool on = (mbits >> e) & 1;
                const float w0 = on ? __expf(fminf(l0[e], 11.f)) : 0.f;
                const float w1v = on ? __expf(fminf(l1[e], 11.f)) : 0.f;
                const float w2v = on ? __expf(fminf(l2[e], 11.f)) : 0.f;
                const float w3 = on ? __expf(fminf(l3[e], 11.f)) : 0.f;
                sm[0] += w0; sm[1] += w1v; sm[2] += w2v; sm[3] += w3;
                a[0][ch * 8 + e] = (_Float16)w0;
                a[1][ch * 8 + e] = (_Float16)w1v;
                a[2][ch * 8 + e] = (_Float16)w2v;
                a[3][ch * 8 + e] = (_Float16)w3;
            }
        }

#if USE_TDM
        __builtin_amdgcn_s_wait_tensorcnt(0);    // our slab is in LDS now
        asm volatile("" ::: "memory");           // TDM wrote LDS behind clang's back
        #pragma unroll
        for (int h = 0; h < HH; ++h) {
            #pragma unroll
            for (int dt = 0; dt < 2; ++dt) {
                const int dim = h * 32 + dt * 16 + nl;
                const v16h bm = load_v16h_g(&vtile[wv][dim][hi * 16]);
                acc[h][dt] = __builtin_amdgcn_wmma_f32_16x16x32_f16(
                    false, a[h], false, bm, (short)0, acc[h][dt], false, false);
            }
        }
#else
        #pragma unroll
        for (int h = 0; h < HH; ++h) {
            #pragma unroll
            for (int dt = 0; dt < 2; ++dt) {
                const int dim = h * 32 + dt * 16 + nl;
                const _Float16* p = Vt + (size_t)dim * ROWS + b * NN + mbase + hi * 16;
                const v16h bm = load_v16h_g(p);
                acc[h][dt] = __builtin_amdgcn_wmma_f32_16x16x32_f16(
                    false, a[h], false, bm, (short)0, acc[h][dt], false, false);
            }
        }
#endif
    }

    __syncthreads();   // all waves done with vtile before accW overlays it

    // deterministic cross-wave reduction through LDS (no float atomics)
    #pragma unroll
    for (int h = 0; h < HH; ++h) {
        sumW[wv][hi][h][nl] = sm[h];
        #pragma unroll
        for (int dt = 0; dt < 2; ++dt)
            #pragma unroll
            for (int r = 0; r < 8; ++r)
                accW[wv][h][dt][hi * 8 + r][nl] = acc[h][dt][r];
    }
    __syncthreads();

    for (int idx = tid; idx < 16 * DD; idx += 128) {
        const int qq = idx >> 7, dim = idx & 127;
        const int h = dim >> 5, dt = (dim >> 4) & 1, cc = dim & 15;
        float num = 0.f, den = 0.f;
        #pragma unroll
        for (int w = 0; w < 4; ++w) {
            num += accW[w][h][dt][qq][cc];
            den += sumW[w][0][h][qq] + sumW[w][1][h][qq];
        }
        attnout[(size_t)(b * NN + q0 + qq) * DD + dim] = (_Float16)(num / den);
    }
}

// ---------------------------------------------------------------------------
// Row-wise LayerNorm + exact GELU. One block per row, one thread per channel.
// ---------------------------------------------------------------------------
__global__ __launch_bounds__(128) void ln_gelu(
    const float* __restrict__ h2, const float* __restrict__ g,
    const float* __restrict__ bta, float* __restrict__ out)
{
    __shared__ float red[128];
    const int row = blockIdx.x, t = threadIdx.x;
    const float v = h2[(size_t)row * DD + t];
    red[t] = v; __syncthreads();
    #pragma unroll
    for (int s = 64; s > 0; s >>= 1) { if (t < s) red[t] += red[t + s]; __syncthreads(); }
    const float mu = red[0] * (1.f / DD);
    __syncthreads();
    const float dv = v - mu;
    red[t] = dv * dv; __syncthreads();
    #pragma unroll
    for (int s = 64; s > 0; s >>= 1) { if (t < s) red[t] += red[t + s]; __syncthreads(); }
    const float var = red[0] * (1.f / DD);
    const float ln  = g[t] * dv * rsqrtf(var + LNEPS) + bta[t];
    out[(size_t)row * DD + t] = 0.5f * ln * (1.f + erff(ln * 0.70710678118654752f));
}

// ---------------------------------------------------------------------------
extern "C" void kernel_launch(void* const* d_in, const int* in_sizes, int n_in,
                              void* d_out, int out_size, void* d_ws, size_t ws_size,
                              hipStream_t stream)
{
    const float* x      = (const float*)d_in[0];
    const float* coords = (const float*)d_in[1];
    const float* rdirs  = (const float*)d_in[2];
    const float* w1     = (const float*)d_in[3];
    const float* b1     = (const float*)d_in[4];
    const float* w2     = (const float*)d_in[5];
    const float* b2     = (const float*)d_in[6];
    const float* wvw    = (const float*)d_in[7];
    const float* bvb    = (const float*)d_in[8];
    const float* wo     = (const float*)d_in[9];
    const float* bo     = (const float*)d_in[10];
    const float* wd     = (const float*)d_in[11];
    const float* bd     = (const float*)d_in[12];
    const float* lng    = (const float*)d_in[13];
    const float* lnb    = (const float*)d_in[14];
    float* out = (float*)d_out;

    char* ws = (char*)d_ws;
    _Float16* Vt   = (_Float16*)(ws + 0);                       // [128][8192] f16
    _Float16* attn = (_Float16*)(ws + 2u * 1024 * 1024);        // [8192][128] f16
    _Float16* outp = (_Float16*)(ws + 4u * 1024 * 1024);        // [8192][128] f16
    float*    h2   = (float*)   (ws + 6u * 1024 * 1024);        // [8192][128] f32

    // K1: V^T = (x @ wv^T + bv)^T, stored dim-major f16 (WMMA-B friendly)
    gemm_wmma<false, true, false, false><<<dim3(1024), dim3(128), 0, stream>>>(
        x, wvw, bvb, nullptr, Vt);
    // K2: fused radius attention -> attnout f16 (TDM-staged V tiles)
    attn_kernel<<<dim3(BB * (NN / 16)), dim3(128), 0, stream>>>(
        coords, rdirs, w1, b1, w2, b2, Vt, attn);
    // K3: out = x + attn @ wo^T + bo  (f16 row-major)
    gemm_wmma<true, false, true, true><<<dim3(1024), dim3(128), 0, stream>>>(
        attn, wo, bo, x, outp);
    // K4: h2 = out @ wd^T + bd  (f32 row-major)
    gemm_wmma<true, false, false, false><<<dim3(1024), dim3(128), 0, stream>>>(
        outp, wd, bd, nullptr, h2);
    // K5: LayerNorm + exact GELU -> d_out
    ln_gelu<<<dim3(ROWS), dim3(128), 0, stream>>>(h2, lng, lnb, out);
}
